// ChunkGNNEncoder_9466107920562
// MI455X (gfx1250) — matrix operand order repaired and checked by reading, not compiled, not run on hardware
//
#include <hip/hip_runtime.h>

typedef __attribute__((ext_vector_type(16))) __bf16 v16bf;
typedef __attribute__((ext_vector_type(8)))  float  v8f;

#define IN_DIM 768
#define HID    256
#define OUTD   128

// ---------------------------------------------------------------------------
// Pack a row-major f32 weight (K x N) into the per-lane B-fragment layout:
// index = ((kt*NTtot + nt)*32 + lane)*16 + i
//   -> W[kt*32 + (lane&16?16:0) + i][nt*16 + (lane&15)]
// ---------------------------------------------------------------------------
__global__ void pack_w_kernel(const float* __restrict__ W, __bf16* __restrict__ out,
                              int K, int N) {
    int t = blockIdx.x * blockDim.x + threadIdx.x;
    int total = K * N;
    if (t >= total) return;
    int i    = t & 15;
    int lane = (t >> 4) & 31;
    int NT   = N >> 4;
    int nt   = (t >> 9) % NT;
    int kt   = t / (512 * NT);
    int k = kt * 32 + ((lane & 16) ? 16 : 0) + i;
    int n = nt * 16 + (lane & 15);
    out[t] = (__bf16)W[(size_t)k * N + n];
}

// ---------------------------------------------------------------------------
// WMMA GEMM: C[M, 16*ntStride] = A[M,K] (f32, lda) * Bpacked (bf16)
// One wave owns a 16-row strip and NT=8 column tiles (64 acc VGPRs -> no spill).
// gridDim.y selects the 128-wide column group: ntBase = blockIdx.y * 8.
// ntStride = N/16 of the packed weight. M % 16 == 0, K % 32 == 0.
// ---------------------------------------------------------------------------
template <int NT>
__global__ void __launch_bounds__(256)
gemm_bf16_wmma(const float* __restrict__ A, int M, int K, int lda,
               const __bf16* __restrict__ Bp, int ntStride,
               float* __restrict__ C, int ldc,
               const float* __restrict__ bias, int relu) {
    int wid  = (blockIdx.x * blockDim.x + threadIdx.x) >> 5;
    int lane = threadIdx.x & 31;
    int rowBase = wid * 16;
    if (rowBase >= M) return;
    int ntBase = blockIdx.y * NT;

    int hiK = (lane & 16) ? 16 : 0;          // A-fragment K sub-offset per half-wave
    int row = rowBase + (lane & 15);
    const float* Ar = A + (size_t)row * lda;

    v8f acc[NT];
    v8f zero = {0.f, 0.f, 0.f, 0.f, 0.f, 0.f, 0.f, 0.f};
#pragma unroll
    for (int nt = 0; nt < NT; ++nt) acc[nt] = zero;

    int KT = K >> 5;
    for (int kt = 0; kt < KT; ++kt) {
        const float* Ak = Ar + kt * 32;
        float4 p0 = *(const float4*)(Ak + hiK);
        float4 p1 = *(const float4*)(Ak + hiK + 4);
        float4 p2 = *(const float4*)(Ak + hiK + 8);
        float4 p3 = *(const float4*)(Ak + hiK + 12);
        v16bf a;
        a[0]  = (__bf16)p0.x; a[1]  = (__bf16)p0.y; a[2]  = (__bf16)p0.z; a[3]  = (__bf16)p0.w;
        a[4]  = (__bf16)p1.x; a[5]  = (__bf16)p1.y; a[6]  = (__bf16)p1.z; a[7]  = (__bf16)p1.w;
        a[8]  = (__bf16)p2.x; a[9]  = (__bf16)p2.y; a[10] = (__bf16)p2.z; a[11] = (__bf16)p2.w;
        a[12] = (__bf16)p3.x; a[13] = (__bf16)p3.y; a[14] = (__bf16)p3.z; a[15] = (__bf16)p3.w;

        const __bf16* Bk = Bp + ((size_t)kt * ntStride + ntBase) * 512 + lane * 16;
#pragma unroll
        for (int nt = 0; nt < NT; ++nt) {
            v16bf b = *(const v16bf*)(Bk + nt * 512);
            acc[nt] = __builtin_amdgcn_wmma_f32_16x16x32_bf16(
                false, a, false, b, (short)0, acc[nt], false, false);
        }
    }

    int hi = (lane >> 4) & 1;
    int lo = lane & 15;
#pragma unroll
    for (int nt = 0; nt < NT; ++nt) {
        int col = (ntBase + nt) * 16 + lo;
        float bv = bias ? bias[col] : 0.f;
#pragma unroll
        for (int r = 0; r < 8; ++r) {
            int orow = rowBase + r + 8 * hi;
            float v = acc[nt][r] + bv;
            if (relu) v = fmaxf(v, 0.f);
            C[(size_t)orow * ldc + col] = v;
        }
    }
}

// ---------------------------------------------------------------------------
// Degree / normalization kernels
// ---------------------------------------------------------------------------
__global__ void set_ones_kernel(float* p, int n) {
    int i = blockIdx.x * blockDim.x + threadIdx.x;
    if (i < n) p[i] = 1.0f;                 // self-loop contribution
}

__global__ void deg_accum_kernel(const int* __restrict__ tgt, float* deg, int E) {
    int e = blockIdx.x * blockDim.x + threadIdx.x;
    if (e < E) unsafeAtomicAdd(&deg[tgt[e]], 1.0f);
}

__global__ void rsqrt_inplace_kernel(float* p, int n) {
    int i = blockIdx.x * blockDim.x + threadIdx.x;
    if (i < n) p[i] = rsqrtf(p[i]);         // deg >= 1 always (self-loops)
}

// agg[i][c] = h[i][c] * dinv[i]^2   (self-loop term; also initializes agg)
__global__ void self_init_kernel(const float* __restrict__ h, const float* __restrict__ dinv,
                                 float* __restrict__ agg, int NN) {
    int t = blockIdx.x * blockDim.x + threadIdx.x;
    int node = t >> 6;
    if (node >= NN) return;
    int c = (t & 63) << 2;
    float di = dinv[node];
    float nrm = di * di;
    float4 hv = *(const float4*)(h + (size_t)node * HID + c);
    float4 o; o.x = hv.x * nrm; o.y = hv.y * nrm; o.z = hv.z * nrm; o.w = hv.w * nrm;
    *(float4*)(agg + (size_t)node * HID + c) = o;
}

// agg[t][c] += h[s][c] * dinv[s]*dinv[t]  over edges
__global__ void edge_scatter_kernel(const int* __restrict__ src, const int* __restrict__ tgt,
                                    const float* __restrict__ dinv,
                                    const float* __restrict__ h, float* __restrict__ agg, int E) {
    int t = blockIdx.x * blockDim.x + threadIdx.x;
    int e = t >> 6;
    if (e >= E) return;
    int c = (t & 63) << 2;
    int s = src[e], d = tgt[e];
    float nrm = dinv[s] * dinv[d];
    float4 hv = *(const float4*)(h + (size_t)s * HID + c);
    float* out = agg + (size_t)d * HID + c;
    unsafeAtomicAdd(out + 0, hv.x * nrm);
    unsafeAtomicAdd(out + 1, hv.y * nrm);
    unsafeAtomicAdd(out + 2, hv.z * nrm);
    unsafeAtomicAdd(out + 3, hv.w * nrm);
}

// in-place: x = relu(x + b)
__global__ void bias_relu_kernel(float* __restrict__ x, const float* __restrict__ b, int NN) {
    int t = blockIdx.x * blockDim.x + threadIdx.x;
    int node = t >> 6;
    if (node >= NN) return;
    int c = (t & 63) << 2;
    float4 bv = *(const float4*)(b + c);
    float4 v  = *(float4*)(x + (size_t)node * HID + c);
    v.x = fmaxf(v.x + bv.x, 0.f); v.y = fmaxf(v.y + bv.y, 0.f);
    v.z = fmaxf(v.z + bv.z, 0.f); v.w = fmaxf(v.w + bv.w, 0.f);
    *(float4*)(x + (size_t)node * HID + c) = v;
}

__global__ void zero_kernel(float* p, int n) {
    int i = blockIdx.x * blockDim.x + threadIdx.x;
    if (i < n) p[i] = 0.f;
}

// segment-sum pooling over batch ids
__global__ void pool_accum_kernel(const float* __restrict__ h, const int* __restrict__ batch,
                                  float* __restrict__ pool, float* __restrict__ cnt, int NN) {
    int t = blockIdx.x * blockDim.x + threadIdx.x;
    int node = t >> 6;
    if (node >= NN) return;
    int c = (t & 63) << 2;
    int g = batch[node];
    float4 hv = *(const float4*)(h + (size_t)node * HID + c);
    float* out = pool + (size_t)g * HID + c;
    unsafeAtomicAdd(out + 0, hv.x);
    unsafeAtomicAdd(out + 1, hv.y);
    unsafeAtomicAdd(out + 2, hv.z);
    unsafeAtomicAdd(out + 3, hv.w);
    if (c == 0) unsafeAtomicAdd(&cnt[g], 1.0f);
}

__global__ void mean_kernel(const float* __restrict__ pool, const float* __restrict__ cnt,
                            float* __restrict__ mean, int total) {
    int i = blockIdx.x * blockDim.x + threadIdx.x;
    if (i >= total) return;
    int g = i >> 8;                          // HID = 256
    mean[i] = pool[i] / fmaxf(cnt[g], 1.0f);
}

// ---------------------------------------------------------------------------
extern "C" void kernel_launch(void* const* d_in, const int* in_sizes, int n_in,
                              void* d_out, int out_size, void* d_ws, size_t ws_size,
                              hipStream_t stream) {
    const float* x    = (const float*)d_in[0];
    const int*   eidx = (const int*)d_in[1];
    const int*   bat  = (const int*)d_in[2];
    const float* W1   = (const float*)d_in[3];
    const float* b1   = (const float*)d_in[4];
    const float* W2   = (const float*)d_in[5];
    const float* b2   = (const float*)d_in[6];
    const float* Wp   = (const float*)d_in[7];
    const float* bp   = (const float*)d_in[8];
    float* out = (float*)d_out;

    const int NN = in_sizes[0] / IN_DIM;     // 50000
    const int E  = in_sizes[1] / 2;          // 300000
    const int G  = out_size / OUTD;          // 64
    const int* src = eidx;
    const int* tgt = eidx + E;

    // ---- carve workspace (256B aligned chunks) ----
    char* p = (char*)d_ws;
    auto carve = [&](size_t bytes) -> void* {
        void* r = (void*)p;
        p += (bytes + 255) & ~(size_t)255;
        return r;
    };
    float*  bufA = (float*)carve((size_t)NN * HID * 4);
    float*  bufB = (float*)carve((size_t)NN * HID * 4);
    float*  dinv = (float*)carve((size_t)NN * 4);
    __bf16* w1p  = (__bf16*)carve((size_t)IN_DIM * HID * 2);
    __bf16* w2p  = (__bf16*)carve((size_t)HID * HID * 2);
    __bf16* wpp  = (__bf16*)carve((size_t)HID * OUTD * 2);
    float*  pool = (float*)carve((size_t)G * HID * 4);
    float*  cnt  = (float*)carve((size_t)G * 4);
    float*  mean = (float*)carve((size_t)G * HID * 4);

    const int BLK = 256;
    auto cdiv = [](int a, int b) { return (a + b - 1) / b; };

    // ---- pack weights to bf16 fragment layout ----
    pack_w_kernel<<<cdiv(IN_DIM * HID, BLK), BLK, 0, stream>>>(W1, w1p, IN_DIM, HID);
    pack_w_kernel<<<cdiv(HID * HID,   BLK), BLK, 0, stream>>>(W2, w2p, HID, HID);
    pack_w_kernel<<<cdiv(HID * OUTD,  BLK), BLK, 0, stream>>>(Wp, wpp, HID, OUTD);

    // ---- degree -> dinv ----
    set_ones_kernel<<<cdiv(NN, BLK), BLK, 0, stream>>>(dinv, NN);
    deg_accum_kernel<<<cdiv(E, BLK), BLK, 0, stream>>>(tgt, dinv, E);
    rsqrt_inplace_kernel<<<cdiv(NN, BLK), BLK, 0, stream>>>(dinv, NN);

    const int gemmBlocksX = cdiv((NN / 16) * 32, BLK);   // one wave per 16-row strip
    const dim3 gemmGrid(gemmBlocksX, HID / 128);         // 2 column halves of 128

    // ---- layer 1: h1 = x @ W1 ; agg ; relu(+b1) ----
    gemm_bf16_wmma<8><<<gemmGrid, BLK, 0, stream>>>(
        x, NN, IN_DIM, IN_DIM, w1p, HID / 16, bufA, HID, nullptr, 0);
    self_init_kernel<<<cdiv(NN * 64, BLK), BLK, 0, stream>>>(bufA, dinv, bufB, NN);
    edge_scatter_kernel<<<cdiv(E * 64, BLK), BLK, 0, stream>>>(src, tgt, dinv, bufA, bufB, E);
    bias_relu_kernel<<<cdiv(NN * 64, BLK), BLK, 0, stream>>>(bufB, b1, NN);

    // ---- layer 2: h2 = act1 @ W2 ; agg ; relu(+b2) ----
    gemm_bf16_wmma<8><<<gemmGrid, BLK, 0, stream>>>(
        bufB, NN, HID, HID, w2p, HID / 16, bufA, HID, nullptr, 0);
    self_init_kernel<<<cdiv(NN * 64, BLK), BLK, 0, stream>>>(bufA, dinv, bufB, NN);
    edge_scatter_kernel<<<cdiv(E * 64, BLK), BLK, 0, stream>>>(src, tgt, dinv, bufA, bufB, E);
    bias_relu_kernel<<<cdiv(NN * 64, BLK), BLK, 0, stream>>>(bufB, b2, NN);

    // ---- global mean pool ----
    zero_kernel<<<cdiv(G * HID, BLK), BLK, 0, stream>>>(pool, G * HID);
    zero_kernel<<<1, BLK, 0, stream>>>(cnt, G);
    pool_accum_kernel<<<cdiv(NN * 64, BLK), BLK, 0, stream>>>(bufB, bat, pool, cnt, NN);
    mean_kernel<<<cdiv(G * HID, BLK), BLK, 0, stream>>>(pool, cnt, mean, G * HID);

    // ---- final projection: out = mean @ Wp + bp  (64x256x128, one N-group) ----
    gemm_bf16_wmma<8><<<dim3(1, 1), (G / 16) * 32, 0, stream>>>(
        mean, G, HID, HID, wpp, OUTD / 16, out, OUTD, bp, 0);
}
